// RNN_15625091023237
// MI455X (gfx1250) — matrix operand order repaired and checked by reading, not compiled, not run on hardware
//
#include <hip/hip_runtime.h>
#include <hip/hip_bf16.h>

// Problem sizes (fixed by reference)
#define B_TOT 1024
#define T_LEN 256
#define VOCAB_SZ 115
#define EMB_SZ 128
#define HID_SZ 512

// Tiling
#define BT 16            // batch rows per workgroup
#define NWG (B_TOT / BT) // 64 workgroups
#define THREADS 512      // 16 waves (wave32); each wave owns 32 h-columns
#define HS 520           // padded LDS row stride for h (bf16 elems)
#define KHALF 256        // K in [256,512) staged in LDS; K in [0,256) in registers
#define WS 264           // padded LDS row stride for staged W (bf16 elems)
#define WLDS_ELEMS (HID_SZ * WS)           // 135168 ushorts = 270336 B
#define HLDS_ELEMS (2 * BT * HS)           // 16640 ushorts = 33280 B

typedef __attribute__((ext_vector_type(16))) __bf16 v16bf;
typedef __attribute__((ext_vector_type(8)))  float  v8f;

struct alignas(16) U4 { unsigned x, y, z, w; };
union FragBF { v16bf v; U4 q[2]; };

#if __has_builtin(__builtin_amdgcn_tanhf)
#define TANHF(x) __builtin_amdgcn_tanhf(x)
#else
#define TANHF(x) tanhf(x)
#endif

__device__ __forceinline__ unsigned short f2bf(float f) {
  unsigned u = __float_as_uint(f);
  u += 0x7FFFu + ((u >> 16) & 1u);   // round-to-nearest-even
  return (unsigned short)(u >> 16);
}
__device__ __forceinline__ float bf2f(unsigned short h) {
  return __uint_as_float(((unsigned)h) << 16);
}

// ---------------------------------------------------------------------------
// Precompute 1: E[v][n] = b_h[n] + sum_e W_emb[v][e] * W_xh[e][n]
// (embed + input-projection GEMM collapses into a 115x512 lookup table)
// ---------------------------------------------------------------------------
__global__ void rnn_precompute_E(const float* __restrict__ W_emb,
                                 const float* __restrict__ W_xh,
                                 const float* __restrict__ b_h,
                                 float* __restrict__ E) {
  int idx = blockIdx.x * blockDim.x + threadIdx.x;
  if (idx >= VOCAB_SZ * HID_SZ) return;
  int v = idx / HID_SZ, n = idx % HID_SZ;
  float s = b_h[n];
#pragma unroll 8
  for (int e = 0; e < EMB_SZ; ++e)
    s += W_emb[v * EMB_SZ + e] * W_xh[e * HID_SZ + n];
  E[idx] = s;
}

// ---------------------------------------------------------------------------
// Precompute 2: WhhT[n][k] = bf16(W_hh[k][n])  (transposed for B fragments)
// ---------------------------------------------------------------------------
__global__ void rnn_transpose_whh(const float* __restrict__ W_hh,
                                  unsigned short* __restrict__ WhhT) {
  int idx = blockIdx.x * blockDim.x + threadIdx.x;
  if (idx >= HID_SZ * HID_SZ) return;
  int n = idx / HID_SZ, k = idx % HID_SZ;
  WhhT[idx] = f2bf(W_hh[k * HID_SZ + n]);
}

// ---------------------------------------------------------------------------
// Main recurrence. Each WG owns BT=16 batch rows for all 256 steps.
//  - W_hh K-half [0,256): register-resident B fragments (128 VGPRs/lane),
//    loaded from L2 exactly once.
//  - W_hh K-half [256,512): staged once into LDS (270 KB, padded stride).
//  - h: bf16 in LDS, double-buffered.
//  - E[token] gather (the fused xh_t + b_h term, used as WMMA C operand)
//    software-pipelined one step ahead so the L2 gather latency hides under
//    the WMMA chain + barrier instead of sitting on the serial path.
// ---------------------------------------------------------------------------
__global__ void __launch_bounds__(THREADS, 1)
rnn_scan_wmma(const int* __restrict__ X,
              const float* __restrict__ E,
              const unsigned short* __restrict__ WhhT,
              const float* __restrict__ W_hy,
              const float* __restrict__ b_y,
              float* __restrict__ out) {
  extern __shared__ unsigned short lds[]; // [WLDS_ELEMS | HLDS_ELEMS]

  const int tid   = threadIdx.x;
  const int lane  = tid & 31;
  const int wave  = tid >> 5;      // 0..15
  const int n0    = wave * 32;     // 32 output cols per wave: tiles n0, n0+16
  const int b0    = blockIdx.x * BT;
  const int lhalf = lane >> 4;     // 0 or 1
  const int l15   = lane & 15;

  // ---- Stage W_hh^T K-half [256,512) into LDS (one-time) ----
  for (int col = tid; col < HID_SZ; col += THREADS) {
    const U4* src = (const U4*)&WhhT[(size_t)col * HID_SZ + KHALF];
    U4* dst = (U4*)&lds[col * WS];
#pragma unroll
    for (int q = 0; q < KHALF / 8; ++q) dst[q] = src[q];
  }
  // ---- Zero-init h buffers ----
  const int HOFF = WLDS_ELEMS;
  for (int i = tid; i < HLDS_ELEMS; i += THREADS) lds[HOFF + i] = 0;

  // ---- Load register-resident B fragments: K chunks 0..7 (K < 256) ----
  // B layout (32x16 bf16): lane col = n + l15, 16 consecutive K from
  // kc*32 + lhalf*16 -> 32 contiguous bytes of a WhhT row.
  FragBF Breg[2][8];
#pragma unroll
  for (int j = 0; j < 2; ++j) {
#pragma unroll
    for (int kc = 0; kc < 8; ++kc) {
      const unsigned short* bp =
          &WhhT[(size_t)(n0 + j * 16 + l15) * HID_SZ + kc * 32 + lhalf * 16];
      Breg[j][kc].q[0] = ((const U4*)bp)[0];
      Breg[j][kc].q[1] = ((const U4*)bp)[1];
    }
  }

  // Token + E-table gather for one timestep into a C-operand register pair.
  // C/D layout: VGPR r, this lane holds rows M = r + lhalf*8, col = n + l15.
  auto gather = [&](int t, v8f* a) {
#pragma unroll
    for (int r = 0; r < 8; ++r) {
      int tk = X[(size_t)(b0 + r + lhalf * 8) * T_LEN + t];
      const float* ep = &E[(size_t)tk * HID_SZ + l15];
      a[0][r] = ep[n0];
      a[1][r] = ep[n0 + 16];
    }
  };

  v8f acc[2];
  gather(0, acc);          // prologue gather for t = 0
  __syncthreads();

  int cur = HOFF;
  int nxt = HOFF + BT * HS;

  for (int t = 0; t < T_LEN; ++t) {
    // Issue next step's gather now; completes under the WMMA chain + barrier.
    v8f accn[2];
    const int tn = (t + 1 < T_LEN) ? (t + 1) : (T_LEN - 1);
    gather(tn, accn);

    // ---- K chunks 0..7: B from registers ----
#pragma unroll
    for (int kc = 0; kc < 8; ++kc) {
      FragBF A;
      const int koff = kc * 32 + lhalf * 16;
      A.q[0] = *(const U4*)&lds[cur + l15 * HS + koff];
      A.q[1] = *(const U4*)&lds[cur + l15 * HS + koff + 8];
      acc[0] = __builtin_amdgcn_wmma_f32_16x16x32_bf16(
          false, A.v, false, Breg[0][kc].v, (short)0, acc[0], false, false);
      acc[1] = __builtin_amdgcn_wmma_f32_16x16x32_bf16(
          false, A.v, false, Breg[1][kc].v, (short)0, acc[1], false, false);
    }

    // ---- K chunks 8..15: B from LDS stage ----
    int wbase = 0;
    asm volatile("" : "+v"(wbase)); // defeat LICM: keep B loads inside t loop
#pragma unroll
    for (int kc = 8; kc < 16; ++kc) {
      FragBF A;
      const int koff = kc * 32 + lhalf * 16;
      A.q[0] = *(const U4*)&lds[cur + l15 * HS + koff];
      A.q[1] = *(const U4*)&lds[cur + l15 * HS + koff + 8];
#pragma unroll
      for (int j = 0; j < 2; ++j) {
        FragBF Bf;
        const int widx =
            wbase + (n0 + j * 16 + l15) * WS + (kc * 32 - KHALF) + lhalf * 16;
        Bf.q[0] = *(const U4*)&lds[widx];
        Bf.q[1] = *(const U4*)&lds[widx + 8];
        acc[j] = __builtin_amdgcn_wmma_f32_16x16x32_bf16(
            false, A.v, false, Bf.v, (short)0, acc[j], false, false);
      }
    }

    // ---- tanh + writeback to the other h buffer ----
#pragma unroll
    for (int j = 0; j < 2; ++j) {
      const int n = n0 + j * 16 + l15;
#pragma unroll
      for (int r = 0; r < 8; ++r) {
        const int row = r + lhalf * 8;
        lds[nxt + row * HS + n] = f2bf(TANHF(acc[j][r]));
      }
    }
    __syncthreads();
    int tmp = cur; cur = nxt; nxt = tmp;

    // Rotate pipelined gather into place for the next step.
    acc[0] = accn[0];
    acc[1] = accn[1];
  }

  // ---- Epilogue: logits = h_T @ W_hy + b_y (tiny GEMM, scalar) ----
  for (int o = tid; o < BT * VOCAB_SZ; o += THREADS) {
    int row = o / VOCAB_SZ, v = o % VOCAB_SZ;
    float s = b_y[v];
    for (int k = 0; k < HID_SZ; ++k)
      s += bf2f(lds[cur + row * HS + k]) * W_hy[(size_t)k * VOCAB_SZ + v];
    out[(size_t)(b0 + row) * VOCAB_SZ + v] = s;
  }
}

// ---------------------------------------------------------------------------
extern "C" void kernel_launch(void* const* d_in, const int* in_sizes, int n_in,
                              void* d_out, int out_size, void* d_ws, size_t ws_size,
                              hipStream_t stream) {
  const int*   X     = (const int*)  d_in[0];
  const float* W_emb = (const float*)d_in[1];
  const float* W_xh  = (const float*)d_in[2];
  const float* W_hh  = (const float*)d_in[3];
  const float* b_h   = (const float*)d_in[4];
  const float* W_hy  = (const float*)d_in[5];
  const float* b_y   = (const float*)d_in[6];
  float* out = (float*)d_out;

  // Workspace layout: E (fp32, 115*512) | WhhT (bf16, 512*512)
  float* E = (float*)d_ws;
  unsigned short* WhhT =
      (unsigned short*)((char*)d_ws + (size_t)VOCAB_SZ * HID_SZ * sizeof(float));

  rnn_precompute_E<<<(VOCAB_SZ * HID_SZ + 255) / 256, 256, 0, stream>>>(
      W_emb, W_xh, b_h, E);
  rnn_transpose_whh<<<(HID_SZ * HID_SZ + 255) / 256, 256, 0, stream>>>(
      W_hh, WhhT);

  size_t shmem = (size_t)(WLDS_ELEMS + HLDS_ELEMS) * sizeof(unsigned short);
  rnn_scan_wmma<<<NWG, THREADS, shmem, stream>>>(X, E, WhhT, W_hy, b_y, out);
}
